// SwinTransformerBlock3D_20624432955806
// MI455X (gfx1250) — matrix operand order repaired and checked by reading, not compiled, not run on hardware
//
#include <hip/hip_runtime.h>
#include <math.h>

typedef __attribute__((ext_vector_type(16))) __bf16 v16bf;
typedef __attribute__((ext_vector_type(8)))  __bf16 v8bf;
typedef __attribute__((ext_vector_type(4)))  __bf16 v4bf;
typedef __attribute__((ext_vector_type(8)))  float  v8f;

#define TOK   100352   // B*D*H*W == NWIN*NSEQ
#define NWIN  1024
#define NSEQ  98
#define CCH   128
#define NHH   4
#define HDD   32
#define NPAD  112      // 98 padded to 7*16
#define KPAD  128      // padded K for P@V

// ---------------------------------------------------------------------------
// WMMA fragment loaders (bf16, 16x16x32), per CDNA5 ISA VGPR layouts.
// A frag: row = lane&15; K-halves {kb..kb+7} and {kb+16..kb+23}, kb=(lane>>4)*8
// B frag: col = lane&15; K-halves {kb..kb+7} and {kb+16..kb+23}, kb=(lane>>4)*8
// ---------------------------------------------------------------------------
static __device__ __forceinline__ v16bf ldfragA(const __bf16* p) {
  v8bf lo = *reinterpret_cast<const v8bf*>(p);
  v8bf hi = *reinterpret_cast<const v8bf*>(p + 16);
  return __builtin_shufflevector(lo, hi, 0,1,2,3,4,5,6,7,8,9,10,11,12,13,14,15);
}
static __device__ __forceinline__ v16bf ldfragB(const __bf16* p) {
  v8bf lo = *reinterpret_cast<const v8bf*>(p);
  v8bf hi = *reinterpret_cast<const v8bf*>(p + 16);
  return __builtin_shufflevector(lo, hi, 0,1,2,3,4,5,6,7,8,9,10,11,12,13,14,15);
}

#define WMMA_BF16(a, b, c) \
  __builtin_amdgcn_wmma_f32_16x16x32_bf16(false, (a), false, (b), (short)0, (c), false, false)

// One wave computes a 32(M) x 64(N) tile: 8 accumulators.
// Per K-step: 4 A-loads + 8 B-loads (b128) feed 8 WMMAs -> 1.5 loads/WMMA,
// with 8 independent chains to hide load latency.
// A: row-major [M,K] bf16 (lda), W: row-major [N,K] bf16 (ldw) -> C = A @ W^T
// acc[h*4+t] holds rows mrow0+h*16.., cols ncol0+t*16..
static __device__ __forceinline__ void wmma_tile32x64(
    const __bf16* __restrict__ A, int lda,
    const __bf16* __restrict__ W, int ldw,
    int K, int mrow0, int ncol0, int lane, v8f* acc)
{
  const __bf16* pa0 = A + (size_t)(mrow0 + (lane & 15)) * lda + ((lane >> 4) << 3);
  const __bf16* pa1 = pa0 + (size_t)16 * lda;
  const int bcol = lane & 15;
  const int bkb  = (lane >> 4) << 3;
  const __bf16* pb = W + (size_t)(ncol0 + bcol) * ldw + bkb;
  const int ldw16 = ldw * 16;
  for (int k = 0; k < K; k += 32) {
    v16bf a0 = ldfragA(pa0 + k);
    v16bf a1 = ldfragA(pa1 + k);
    v16bf b0 = ldfragB(pb + k);
    v16bf b1 = ldfragB(pb + k + ldw16);
    v16bf b2 = ldfragB(pb + k + 2 * ldw16);
    v16bf b3 = ldfragB(pb + k + 3 * ldw16);
    acc[0] = WMMA_BF16(a0, b0, acc[0]);
    acc[1] = WMMA_BF16(a0, b1, acc[1]);
    acc[2] = WMMA_BF16(a0, b2, acc[2]);
    acc[3] = WMMA_BF16(a0, b3, acc[3]);
    acc[4] = WMMA_BF16(a1, b0, acc[4]);
    acc[5] = WMMA_BF16(a1, b1, acc[5]);
    acc[6] = WMMA_BF16(a1, b2, acc[6]);
    acc[7] = WMMA_BF16(a1, b3, acc[7]);
  }
}

static __device__ __forceinline__ float wave_sum32(float v) {
#pragma unroll
  for (int m = 16; m >= 1; m >>= 1) v += __shfl_xor(v, m, 32);
  return v;
}

// window token -> original spatial offset (handles cyclic shift (1,3,3))
static __device__ __forceinline__ size_t win_to_spatial(int win, int n, int c) {
  int bb  = win >> 8;          // batch
  int rem = win & 255;
  int id  = rem >> 6;          // D/2 blocks
  int r2  = rem & 63;
  int ih  = r2 >> 3;           // H/7 blocks
  int iw  = r2 & 7;            // W/7 blocks
  int nd  = n / 49, r3 = n % 49, nh = r3 / 7, nw = r3 % 7;
  int od  = (id * 2 + nd + 1) & 7;        // + shift_d, mod 8
  int oh  = (ih * 7 + nh + 3) % 56;       // + shift_h
  int ow  = (iw * 7 + nw + 3) % 56;       // + shift_w
  return ((((size_t)(bb * 8 + od) * 56 + oh) * 56 + ow) * CCH) + c;
}

// ---------------------------------------------------------------------------
// 0) fp32 -> bf16 convert
// ---------------------------------------------------------------------------
__global__ void cvt_bf16_kernel(const float* __restrict__ in,
                                __bf16* __restrict__ outp, int n) {
  int i = blockIdx.x * 256 + threadIdx.x;
  if (i < n) outp[i] = (__bf16)in[i];
}

// 0b) relative position bias table: (NH, N, N) fp32
__global__ void bias_table_kernel(const float* __restrict__ rpb,
                                  const int* __restrict__ ridx,
                                  float* __restrict__ bias_t) {
  int i = blockIdx.x * 256 + threadIdx.x;
  if (i >= NHH * NSEQ * NSEQ) return;
  int h = i / (NSEQ * NSEQ);
  int nm = i % (NSEQ * NSEQ);
  bias_t[i] = rpb[ridx[nm] * NHH + h];
}

// ---------------------------------------------------------------------------
// 1) LN1 + cyclic shift + window partition -> xw bf16 [TOK, 128]
//    one wave per token, float4 per lane
// ---------------------------------------------------------------------------
__global__ __launch_bounds__(256) void ln1_shift_kernel(
    const float* __restrict__ x, const float* __restrict__ g,
    const float* __restrict__ b, __bf16* __restrict__ xw)
{
  int token = blockIdx.x * 8 + (threadIdx.x >> 5);
  if (token >= TOK) return;
  int lane = threadIdx.x & 31;
  int win = token / NSEQ, n = token % NSEQ;
  const float* src = x + win_to_spatial(win, n, 0);
  float4 v = reinterpret_cast<const float4*>(src)[lane];
  float s  = wave_sum32(v.x + v.y + v.z + v.w);
  float sq = wave_sum32(v.x * v.x + v.y * v.y + v.z * v.z + v.w * v.w);
  float mean = s * (1.0f / CCH);
  float var  = sq * (1.0f / CCH) - mean * mean;
  float inv  = rsqrtf(var + 1e-5f);
  int c = lane * 4;
  v4bf r;
  r[0] = (__bf16)((v.x - mean) * inv * g[c + 0] + b[c + 0]);
  r[1] = (__bf16)((v.y - mean) * inv * g[c + 1] + b[c + 1]);
  r[2] = (__bf16)((v.z - mean) * inv * g[c + 2] + b[c + 2]);
  r[3] = (__bf16)((v.w - mean) * inv * g[c + 3] + b[c + 3]);
  reinterpret_cast<v4bf*>(xw + (size_t)token * CCH)[lane] = r;
}

// ---------------------------------------------------------------------------
// 2) QKV GEMM [TOK,128] x [384,128]^T, epilogue scatters q/k/v panels
//    q scaled by HD^-0.5 before bf16 conversion
// ---------------------------------------------------------------------------
__global__ __launch_bounds__(256) void qkv_gemm_kernel(
    const __bf16* __restrict__ xw, const __bf16* __restrict__ wqkv,
    const float* __restrict__ qkv_b,
    __bf16* __restrict__ q, __bf16* __restrict__ kmat, __bf16* __restrict__ v)
{
  int wv = threadIdx.x >> 5, lane = threadIdx.x & 31;
  int task = blockIdx.x * 8 + wv;               // (TOK/32) * 6 tasks
  const int NT = 384 / 64;
  int mt = task / NT, nt = task % NT;
  if (mt >= TOK / 32) return;
  v8f acc[8] = {};
  wmma_tile32x64(xw, CCH, wqkv, CCH, CCH, mt * 32, nt * 64, lane, acc);
#pragma unroll
  for (int hh2 = 0; hh2 < 2; ++hh2) {
    int rbase = mt * 32 + hh2 * 16 + ((lane >> 4) << 3);
#pragma unroll
    for (int t = 0; t < 4; ++t) {
      int col = nt * 64 + t * 16 + (lane & 15);
      int which = col >> 7, hh = (col >> 5) & 3, d = col & 31;
      float bb = qkv_b[col];
      float sc = (which == 0) ? 0.17677669529663689f : 1.0f;  // 32^-0.5
      __bf16* dst = (which == 0) ? q : ((which == 1) ? kmat : v);
#pragma unroll
      for (int i = 0; i < 8; ++i) {
        int row = rbase + i;
        int win = row / NSEQ, n = row % NSEQ;
        float val = (acc[hh2 * 4 + t][i] + bb) * sc;
        dst[((size_t)(win * NHH + hh) * NSEQ + n) * HDD + d] = (__bf16)val;
      }
    }
  }
}

// ---------------------------------------------------------------------------
// 3) Attention: one block per (window, head). LDS-staged q/k/vT + S buffer.
//    S = qk^T + bias + mask -> softmax -> O = P @ v -> attnout bf16 [TOK,128]
// ---------------------------------------------------------------------------
#define SMEM_ATTN (NPAD*HDD*2 + NPAD*HDD*2 + HDD*KPAD*2 + NPAD*KPAD*4)

__global__ __launch_bounds__(256) void attn_kernel(
    const __bf16* __restrict__ q, const __bf16* __restrict__ kmat,
    const __bf16* __restrict__ v, const float* __restrict__ bias_t,
    const float* __restrict__ mask, __bf16* __restrict__ attnout)
{
  extern __shared__ __align__(128) char smem[];
  __bf16* qs  = (__bf16*)smem;                        // [112][32]
  __bf16* ks  = qs + NPAD * HDD;                      // [112][32]
  __bf16* vsT = ks + NPAD * HDD;                      // [32][128] transposed
  float*  S   = (float*)(smem + (NPAD*HDD*2)*2 + HDD*KPAD*2);  // [112][128]

  const int tid = threadIdx.x;
  const int w = blockIdx.x >> 2, h = blockIdx.x & 3;
  const size_t base = (size_t)(w * NHH + h) * NSEQ * HDD;   // 3136 elems
  const unsigned int* qg = (const unsigned int*)(q + base);
  const unsigned int* kg = (const unsigned int*)(kmat + base);
  const __bf16* vg = v + base;

  // phase 0: stage q/k (zero-pad rows 98..111), zero vsT
  unsigned int* qs32 = (unsigned int*)qs;
  unsigned int* ks32 = (unsigned int*)ks;
  unsigned int* vt32 = (unsigned int*)vsT;
  for (int i = tid; i < NPAD * HDD / 2; i += 256) {   // 1792 dwords
    unsigned int qv = (i < NSEQ * HDD / 2) ? qg[i] : 0u;
    unsigned int kv = (i < NSEQ * HDD / 2) ? kg[i] : 0u;
    qs32[i] = qv; ks32[i] = kv;
  }
  for (int i = tid; i < HDD * KPAD / 2; i += 256) vt32[i] = 0u;
  __syncthreads();

  // phase 1: scatter v transposed; compute S tiles
  for (int i = tid; i < NSEQ * HDD; i += 256) {
    int m = i >> 5, d = i & 31;
    vsT[d * KPAD + m] = vg[i];
  }
  const int wv = tid >> 5, lane = tid & 31;
  const int akb = (lane >> 4) << 3;
  const int bkb = (lane >> 4) << 3;
  for (int t = wv; t < 49; t += 8) {
    int mt = t / 7, nt = t % 7;
    const __bf16* pa = qs + (mt * 16 + (lane & 15)) * HDD + akb;
    const __bf16* pb = ks + (nt * 16 + (lane & 15)) * HDD + bkb;
    v16bf a = ldfragA(pa);
    v16bf b = ldfragB(pb);
    v8f c = {};
    c = WMMA_BF16(a, b, c);
    int coln = nt * 16 + (lane & 15);
    int rb = mt * 16 + ((lane >> 4) << 3);
#pragma unroll
    for (int i = 0; i < 8; ++i) {
      int rown = rb + i;
      float add = 0.0f;
      if (rown < NSEQ && coln < NSEQ)
        add = bias_t[((size_t)h * NSEQ + rown) * NSEQ + coln]
            + mask[((size_t)(w & 255) * NSEQ + rown) * NSEQ + coln];
      S[rown * KPAD + coln] = c[i] + add;
    }
  }
  __syncthreads();

  // phase 2: row softmax (rows >= 98 and cols >= 98 zeroed)
  if (tid < NPAD) {
    float* Sr = S + tid * KPAD;
    if (tid < NSEQ) {
      float mx = -1e30f;
      for (int m = 0; m < NSEQ; ++m) mx = fmaxf(mx, Sr[m]);
      float sum = 0.0f;
      for (int m = 0; m < NSEQ; ++m) { float e = __expf(Sr[m] - mx); Sr[m] = e; sum += e; }
      float inv = 1.0f / sum;
      for (int m = 0; m < NSEQ; ++m) Sr[m] *= inv;
      for (int m = NSEQ; m < KPAD; ++m) Sr[m] = 0.0f;
    } else {
      for (int m = 0; m < KPAD; ++m) Sr[m] = 0.0f;
    }
  }
  __syncthreads();

  // phase 3: O = P @ v  (P from fp32 LDS, converted per-fragment; vT contiguous)
  for (int t = wv; t < 14; t += 8) {
    int mt = t >> 1, nt = t & 1;
    int arow = mt * 16 + (lane & 15);
    int bcol = nt * 16 + (lane & 15);
    v8f c = {};
    for (int k = 0; k < KPAD; k += 32) {
      const float* pa = S + arow * KPAD + k + akb;
      v16bf a;
#pragma unroll
      for (int j = 0; j < 8; ++j) {
        a[j]     = (__bf16)pa[j];
        a[8 + j] = (__bf16)pa[16 + j];
      }
      v16bf b = ldfragB(vsT + bcol * KPAD + k + bkb);
      c = WMMA_BF16(a, b, c);
    }
    int rb = mt * 16 + ((lane >> 4) << 3);
    int ch = h * HDD + nt * 16 + (lane & 15);
#pragma unroll
    for (int i = 0; i < 8; ++i) {
      int n = rb + i;
      if (n < NSEQ)
        attnout[((size_t)w * NSEQ + n) * CCH + ch] = (__bf16)c[i];
    }
  }
}

// ---------------------------------------------------------------------------
// 4) proj GEMM + window reverse + un-shift + residual -> x1 fp32 [TOK,128]
// ---------------------------------------------------------------------------
__global__ __launch_bounds__(256) void proj_gemm_kernel(
    const __bf16* __restrict__ ao, const __bf16* __restrict__ wproj,
    const float* __restrict__ proj_b, const float* __restrict__ xin,
    float* __restrict__ x1)
{
  int wv = threadIdx.x >> 5, lane = threadIdx.x & 31;
  int task = blockIdx.x * 8 + wv;
  int mt = task >> 1, nt = task & 1;
  if (mt >= TOK / 32) return;
  v8f acc[8] = {};
  wmma_tile32x64(ao, CCH, wproj, CCH, CCH, mt * 32, nt * 64, lane, acc);
#pragma unroll
  for (int hh2 = 0; hh2 < 2; ++hh2) {
    int rbase = mt * 32 + hh2 * 16 + ((lane >> 4) << 3);
#pragma unroll
    for (int t = 0; t < 4; ++t) {
      int col = nt * 64 + t * 16 + (lane & 15);
      float bb = proj_b[col];
#pragma unroll
      for (int i = 0; i < 8; ++i) {
        int row = rbase + i;
        int win = row / NSEQ, n = row % NSEQ;
        size_t idx = win_to_spatial(win, n, col);
        x1[idx] = xin[idx] + acc[hh2 * 4 + t][i] + bb;
      }
    }
  }
}

// ---------------------------------------------------------------------------
// 5) LN2 -> bf16 h_in [TOK,128]
// ---------------------------------------------------------------------------
__global__ __launch_bounds__(256) void ln2_kernel(
    const float* __restrict__ x1, const float* __restrict__ g,
    const float* __restrict__ b, __bf16* __restrict__ hin)
{
  int token = blockIdx.x * 8 + (threadIdx.x >> 5);
  if (token >= TOK) return;
  int lane = threadIdx.x & 31;
  const float* src = x1 + (size_t)token * CCH;
  float4 v = reinterpret_cast<const float4*>(src)[lane];
  float s  = wave_sum32(v.x + v.y + v.z + v.w);
  float sq = wave_sum32(v.x * v.x + v.y * v.y + v.z * v.z + v.w * v.w);
  float mean = s * (1.0f / CCH);
  float var  = sq * (1.0f / CCH) - mean * mean;
  float inv  = rsqrtf(var + 1e-5f);
  int c = lane * 4;
  v4bf r;
  r[0] = (__bf16)((v.x - mean) * inv * g[c + 0] + b[c + 0]);
  r[1] = (__bf16)((v.y - mean) * inv * g[c + 1] + b[c + 1]);
  r[2] = (__bf16)((v.z - mean) * inv * g[c + 2] + b[c + 2]);
  r[3] = (__bf16)((v.w - mean) * inv * g[c + 3] + b[c + 3]);
  reinterpret_cast<v4bf*>(hin + (size_t)token * CCH)[lane] = r;
}

// ---------------------------------------------------------------------------
// 6) fc1 GEMM [TOK,128]x[512,128]^T + exact GELU -> h1 bf16 [TOK,512]
// ---------------------------------------------------------------------------
__global__ __launch_bounds__(256) void fc1_gemm_kernel(
    const __bf16* __restrict__ hin, const __bf16* __restrict__ wfc1,
    const float* __restrict__ fc1_b, __bf16* __restrict__ h1)
{
  int wv = threadIdx.x >> 5, lane = threadIdx.x & 31;
  int task = blockIdx.x * 8 + wv;
  const int NT = 512 / 64;
  int mt = task / NT, nt = task % NT;
  if (mt >= TOK / 32) return;
  v8f acc[8] = {};
  wmma_tile32x64(hin, CCH, wfc1, CCH, CCH, mt * 32, nt * 64, lane, acc);
#pragma unroll
  for (int hh2 = 0; hh2 < 2; ++hh2) {
    int rbase = mt * 32 + hh2 * 16 + ((lane >> 4) << 3);
#pragma unroll
    for (int t = 0; t < 4; ++t) {
      int col = nt * 64 + t * 16 + (lane & 15);
      float bb = fc1_b[col];
#pragma unroll
      for (int i = 0; i < 8; ++i) {
        float val = acc[hh2 * 4 + t][i] + bb;
        float gl = 0.5f * val * (1.0f + erff(val * 0.7071067811865476f));
        h1[(size_t)(rbase + i) * 512 + col] = (__bf16)gl;
      }
    }
  }
}

// ---------------------------------------------------------------------------
// 7) fc2 GEMM [TOK,512]x[128,512]^T + bias + residual -> out fp32 [TOK,128]
// ---------------------------------------------------------------------------
__global__ __launch_bounds__(256) void fc2_gemm_kernel(
    const __bf16* __restrict__ h1, const __bf16* __restrict__ wfc2,
    const float* __restrict__ fc2_b, const float* __restrict__ x1,
    float* __restrict__ outp)
{
  int wv = threadIdx.x >> 5, lane = threadIdx.x & 31;
  int task = blockIdx.x * 8 + wv;
  int mt = task >> 1, nt = task & 1;
  if (mt >= TOK / 32) return;
  v8f acc[8] = {};
  wmma_tile32x64(h1, 512, wfc2, 512, 512, mt * 32, nt * 64, lane, acc);
#pragma unroll
  for (int hh2 = 0; hh2 < 2; ++hh2) {
    int rbase = mt * 32 + hh2 * 16 + ((lane >> 4) << 3);
#pragma unroll
    for (int t = 0; t < 4; ++t) {
      int col = nt * 64 + t * 16 + (lane & 15);
      float bb = fc2_b[col];
#pragma unroll
      for (int i = 0; i < 8; ++i) {
        size_t idx = (size_t)(rbase + i) * CCH + col;
        outp[idx] = x1[idx] + acc[hh2 * 4 + t][i] + bb;
      }
    }
  }
}

// ---------------------------------------------------------------------------
extern "C" void kernel_launch(void* const* d_in, const int* in_sizes, int n_in,
                              void* d_out, int out_size, void* d_ws, size_t ws_size,
                              hipStream_t stream) {
  (void)in_sizes; (void)n_in; (void)out_size; (void)ws_size;
  const float* x      = (const float*)d_in[0];
  const float* mask   = (const float*)d_in[1];
  const int*   ridx   = (const int*)d_in[2];
  const float* n1g    = (const float*)d_in[3];
  const float* n1b    = (const float*)d_in[4];
  const float* qkv_w  = (const float*)d_in[5];
  const float* qkv_b  = (const float*)d_in[6];
  const float* rpb    = (const float*)d_in[7];
  const float* proj_w = (const float*)d_in[8];
  const float* proj_b = (const float*)d_in[9];
  const float* n2g    = (const float*)d_in[10];
  const float* n2b    = (const float*)d_in[11];
  const float* fc1_w  = (const float*)d_in[12];
  const float* fc1_b  = (const float*)d_in[13];
  const float* fc2_w  = (const float*)d_in[14];
  const float* fc2_b  = (const float*)d_in[15];
  float* outp = (float*)d_out;

  char* ws = (char*)d_ws;
  size_t o = 0;
  auto alloc = [&](size_t bytes) {
    o = (o + 255) & ~(size_t)255;
    size_t r = o; o += bytes; return r;
  };
  size_t off_wqkv = alloc((size_t)384 * 128 * 2);
  size_t off_wprj = alloc((size_t)128 * 128 * 2);
  size_t off_wfc1 = alloc((size_t)512 * 128 * 2);
  size_t off_wfc2 = alloc((size_t)128 * 512 * 2);
  size_t off_bias = alloc((size_t)NHH * NSEQ * NSEQ * 4);
  size_t off_x1   = alloc((size_t)TOK * CCH * 4);
  size_t off_B    = alloc((size_t)TOK * CCH * 2);   // xw -> attnout -> h_in
  size_t off_A    = alloc((size_t)TOK * 512 * 2);   // q|k|v -> h1

  __bf16* wqkv = (__bf16*)(ws + off_wqkv);
  __bf16* wprj = (__bf16*)(ws + off_wprj);
  __bf16* wfc1 = (__bf16*)(ws + off_wfc1);
  __bf16* wfc2 = (__bf16*)(ws + off_wfc2);
  float*  bias_t = (float*)(ws + off_bias);
  float*  x1   = (float*)(ws + off_x1);
  __bf16* bufB = (__bf16*)(ws + off_B);             // 25.7 MB
  __bf16* bufA = (__bf16*)(ws + off_A);             // 102.8 MB
  __bf16* qb = bufA;
  __bf16* kb = bufA + (size_t)TOK * CCH;
  __bf16* vb = bufA + (size_t)2 * TOK * CCH;

  // 0) weight converts + bias table
  cvt_bf16_kernel<<<(384 * 128 + 255) / 256, 256, 0, stream>>>(qkv_w, wqkv, 384 * 128);
  cvt_bf16_kernel<<<(128 * 128 + 255) / 256, 256, 0, stream>>>(proj_w, wprj, 128 * 128);
  cvt_bf16_kernel<<<(512 * 128 + 255) / 256, 256, 0, stream>>>(fc1_w, wfc1, 512 * 128);
  cvt_bf16_kernel<<<(128 * 512 + 255) / 256, 256, 0, stream>>>(fc2_w, wfc2, 128 * 512);
  bias_table_kernel<<<(NHH * NSEQ * NSEQ + 255) / 256, 256, 0, stream>>>(rpb, ridx, bias_t);

  // 1) LN1 + shift + partition -> xw (bufB)
  ln1_shift_kernel<<<(TOK + 7) / 8, 256, 0, stream>>>(x, n1g, n1b, bufB);

  // 2) QKV GEMM (wave tile 32x64)
  qkv_gemm_kernel<<<(TOK / 32) * 6 / 8, 256, 0, stream>>>(bufB, wqkv, qkv_b, qb, kb, vb);

  // 3) attention -> attnout (bufB, xw no longer needed)
  attn_kernel<<<NWIN * NHH, 256, SMEM_ATTN, stream>>>(qb, kb, vb, bias_t, mask, bufB);

  // 4) proj + reverse + residual -> x1
  proj_gemm_kernel<<<(TOK / 32) * 2 / 8, 256, 0, stream>>>(bufB, wprj, proj_b, x, x1);

  // 5) LN2 -> h_in (bufB, attnout no longer needed)
  ln2_kernel<<<(TOK + 7) / 8, 256, 0, stream>>>(x1, n2g, n2b, bufB);

  // 6) fc1 + GELU -> h1 (bufA, q/k/v no longer needed)
  fc1_gemm_kernel<<<(TOK / 32) * 8 / 8, 256, 0, stream>>>(bufB, wfc1, fc1_b, bufA);

  // 7) fc2 + residual -> out
  fc2_gemm_kernel<<<(TOK / 32) * 2 / 8, 256, 0, stream>>>(bufA, wfc2, fc2_b, x1, outp);
}
